// FallbackMamba_33552284516399
// MI455X (gfx1250) — matrix operand order repaired and checked by reading, not compiled, not run on hardware
//
#include <hip/hip_runtime.h>
#include <stddef.h>

// ---------------------------------------------------------------------------
// Types for CDNA5 WMMA (wave32, gfx1250)
// ---------------------------------------------------------------------------
typedef __attribute__((ext_vector_type(16))) __bf16   v16bf;
typedef __attribute__((ext_vector_type(8)))  float    v8f;
typedef __attribute__((ext_vector_type(4)))  unsigned int u32x4;

union Frag {                 // 32 bytes: one bf16 A/B operand fragment
    v16bf v;
    u32x4 q[2];
};

__device__ __forceinline__ unsigned short f2bf(float f) {
    unsigned int u = __float_as_uint(f);
    unsigned int r = u + 0x7FFFu + ((u >> 16) & 1u);   // round-to-nearest-even
    return (unsigned short)(r >> 16);
}
__device__ __forceinline__ float bf2f(unsigned short h) {
    return __uint_as_float(((unsigned int)h) << 16);
}

// 16B async copy global -> LDS (ASYNCcnt-tracked, bypasses VGPRs)
__device__ __forceinline__ void async_cp16(unsigned lds_off, const void* gptr) {
    asm volatile("global_load_async_to_lds_b128 %0, %1, off"
                 :: "v"(lds_off), "v"(gptr) : "memory");
}

// Problem constants
#define DM   1024     // d_model
#define EE   2048     // expanded dim
#define NN2E 4096     // 2*E
#define BB   4
#define LL   4096
#define MTOT (BB * LL)   // 16384 rows

// ---------------------------------------------------------------------------
// bf16 WMMA GEMM with async double-buffered LDS staging:
//   D[M,N] = A[M,K] * Wt[N,K]^T + bias     (A bf16 row-major, Wt bf16 [N][K])
//   Block tile 128x128, K-step 32, 256 threads = 8 wave32 (2x4 wave grid),
//   each wave owns a 64x32 sub-tile = 4x2 WMMA accumulators.
//   Tiles for step i+1 stream via GLOBAL_LOAD_ASYNC_TO_LDS_B128 while step i
//   runs on the matrix pipe; s_wait_asynccnt 4 + barrier separates buffers.
// ---------------------------------------------------------------------------
#define BM 128
#define BN 128
#define BK 32
#define PADK 40   // padded LDS row (ushorts): 80B rows, 16B aligned, bank-spread

template <bool OUT_BF16>
__global__ __launch_bounds__(256)
void gemm_bf16_wmma_async(const unsigned short* __restrict__ A,
                          const unsigned short* __restrict__ Wt,  // [N][K]
                          const float* __restrict__ bias,
                          void* __restrict__ Dptr,
                          int K, int lda, int ldd)
{
    __shared__ __align__(16) unsigned short sA[2][BM][PADK];
    __shared__ __align__(16) unsigned short sB[2][BN][PADK];

    const int tid   = threadIdx.x;
    const int lane  = tid & 31;
    const int lmod  = lane & 15;
    const int khalf = lane >> 4;
    const int wave  = tid >> 5;      // 0..7
    const int waveM = wave >> 2;     // 0..1
    const int waveN = wave & 3;      // 0..3
    const int rowBase = blockIdx.x * BM;
    const int colBase = blockIdx.y * BN;

    // This thread's 2 copy slots (of 512 16B-chunks per matrix tile):
    //   p -> row r = p>>2, 16B chunk c = p&3  (64B of K per row)
    const int r0 = tid >> 2,            c0 = tid & 3;
    const int r1 = (tid + 256) >> 2,    c1 = (tid + 256) & 3;

    auto issue_tile = [&](int k0, int buf) {
        const unsigned short* ga0 = A  + (size_t)(rowBase + r0) * lda + k0 + c0 * 8;
        const unsigned short* ga1 = A  + (size_t)(rowBase + r1) * lda + k0 + c1 * 8;
        const unsigned short* gb0 = Wt + (size_t)(colBase + r0) * K   + k0 + c0 * 8;
        const unsigned short* gb1 = Wt + (size_t)(colBase + r1) * K   + k0 + c1 * 8;
        async_cp16((unsigned)(size_t)&sA[buf][r0][c0 * 8], ga0);
        async_cp16((unsigned)(size_t)&sA[buf][r1][c1 * 8], ga1);
        async_cp16((unsigned)(size_t)&sB[buf][r0][c0 * 8], gb0);
        async_cp16((unsigned)(size_t)&sB[buf][r1][c1 * 8], gb1);
    };

    v8f acc[4][2];
    #pragma unroll
    for (int mi = 0; mi < 4; ++mi)
        #pragma unroll
        for (int ni = 0; ni < 2; ++ni)
            acc[mi][ni] = (v8f){};

    const int nK = K / BK;
    issue_tile(0, 0);                                 // prologue: tile 0

    for (int i = 0; i < nK; ++i) {
        const int buf = i & 1;
        if (i + 1 < nK) {
            issue_tile((i + 1) * BK, buf ^ 1);        // stream next tile
            asm volatile("s_wait_asynccnt 0x4" ::: "memory");  // tile i done
        } else {
            asm volatile("s_wait_asynccnt 0x0" ::: "memory");
        }
        __syncthreads();                              // all waves' copies visible

        // ---- B fragments: lane -> N=lmod, elements i -> K = khalf*16 + i ----
        Frag bf[2];
        #pragma unroll
        for (int ni = 0; ni < 2; ++ni) {
            const unsigned short* p = &sB[buf][waveN * 32 + ni * 16 + lmod][khalf * 16];
            bf[ni].q[0] = *(const u32x4*)(p);
            bf[ni].q[1] = *(const u32x4*)(p + 8);
        }
        // ---- A fragments + WMMA ----
        #pragma unroll
        for (int mi = 0; mi < 4; ++mi) {
            Frag af;
            const unsigned short* pa = &sA[buf][waveM * 64 + mi * 16 + lmod][khalf * 16];
            af.q[0] = *(const u32x4*)(pa);        // K = khalf*16 + 0..7
            af.q[1] = *(const u32x4*)(pa + 8);    // K = khalf*16 + 8..15
            #pragma unroll
            for (int ni = 0; ni < 2; ++ni)
                acc[mi][ni] = __builtin_amdgcn_wmma_f32_16x16x32_bf16(
                    false, af.v, false, bf[ni].v, (short)0, acc[mi][ni], false, false);
        }
        __syncthreads();                              // reads done before reuse
    }

    // ---- epilogue: D layout VGPR r -> M = r + 8*khalf, N = lmod ----
    #pragma unroll
    for (int mi = 0; mi < 4; ++mi)
        #pragma unroll
        for (int ni = 0; ni < 2; ++ni) {
            int col = colBase + waveN * 32 + ni * 16 + lmod;
            float bv = bias ? bias[col] : 0.0f;
            #pragma unroll
            for (int r = 0; r < 8; ++r) {
                int row = rowBase + waveM * 64 + mi * 16 + r + khalf * 8;
                float v = acc[mi][ni][r] + bv;
                if (OUT_BF16)
                    ((unsigned short*)Dptr)[(size_t)row * ldd + col] = f2bf(v);
                else
                    ((float*)Dptr)[(size_t)row * ldd + col] = v;
            }
        }
}

// ---------------------------------------------------------------------------
// Prepass: fp32 -> bf16 elementwise convert
// ---------------------------------------------------------------------------
__global__ __launch_bounds__(256)
void convert_bf16_kernel(const float* __restrict__ src,
                         unsigned short* __restrict__ dst)
{
    size_t i = (size_t)blockIdx.x * 256 + threadIdx.x;
    dst[i] = f2bf(src[i]);
}

// ---------------------------------------------------------------------------
// Prepass: W[K][N] fp32 -> Wt[N][K] bf16 (32x32 LDS tile transpose)
// ---------------------------------------------------------------------------
__global__ __launch_bounds__(256)
void transpose_bf16_kernel(const float* __restrict__ W,
                           unsigned short* __restrict__ Wt, int K, int N)
{
    __shared__ float tile[32][33];
    int k0 = blockIdx.x * 32, n0 = blockIdx.y * 32;
    #pragma unroll
    for (int j = threadIdx.y; j < 32; j += 8)
        tile[j][threadIdx.x] = W[(size_t)(k0 + j) * N + n0 + threadIdx.x];
    __syncthreads();
    #pragma unroll
    for (int j = threadIdx.y; j < 32; j += 8)
        Wt[(size_t)(n0 + j) * K + k0 + threadIdx.x] = f2bf(tile[threadIdx.x][j]);
}

// ---------------------------------------------------------------------------
// Causal depthwise conv1d (D_CONV=4) over the x_proj half of xz (bf16 in/out)
// ---------------------------------------------------------------------------
__global__ __launch_bounds__(256)
void conv_kernel(const unsigned short* __restrict__ xz,
                 const float* __restrict__ cw,   // [E][1][4]
                 const float* __restrict__ cb,   // [E]
                 unsigned short* __restrict__ xc)
{
    size_t idx = (size_t)blockIdx.x * 256 + threadIdx.x;   // MTOT*EE threads
    int e = (int)(idx & (EE - 1));
    size_t row = idx >> 11;          // b*L + t
    int t = (int)(row & (LL - 1));
    const unsigned short* xp = xz + row * NN2E + e;        // x_proj column e
    float a = cb[e];
    #pragma unroll
    for (int k = 0; k < 4; ++k) {
        int dt = t - 3 + k;
        if (dt >= 0)
            a += bf2f(xp[(ptrdiff_t)(k - 3) * NN2E]) * cw[e * 4 + k];
    }
    xc[idx] = f2bf(a);
}

// ---------------------------------------------------------------------------
// U[M,16] = x_conv[M,2048] @ A[2048,16]  (bf16 WMMA, N=16 = one tile column)
// A is staged transposed in LDS in 512-K chunks. 8 wave32 / block, 128 rows.
// ---------------------------------------------------------------------------
#define SKC 512
__global__ __launch_bounds__(256)
void xA_gemm_wmma(const unsigned short* __restrict__ xc,
                  const float* __restrict__ Amat,
                  float* __restrict__ U)
{
    __shared__ __align__(16) unsigned short sAb[16][SKC];   // [n][k], 16 KB

    const int tid   = threadIdx.x;
    const int lane  = tid & 31;
    const int lmod  = lane & 15;
    const int khalf = lane >> 4;
    const int rowBase = blockIdx.x * 128 + (tid >> 5) * 16;

    v8f acc = (v8f){};

    for (int kc = 0; kc < EE; kc += SKC) {
        #pragma unroll
        for (int i = 0; i < 32; ++i) {                       // 16*512 = 8192 elems
            int idx = tid + i * 256;
            int kk = idx >> 4, n = idx & 15;
            sAb[n][kk] = f2bf(Amat[(size_t)(kc + kk) * 16 + n]);
        }
        __syncthreads();

        for (int ks = 0; ks < SKC; ks += 32) {
            Frag af, bfr;
            const unsigned short* pa =
                xc + (size_t)(rowBase + lmod) * EE + kc + ks + khalf * 16;
            af.q[0] = *(const u32x4*)(pa);
            af.q[1] = *(const u32x4*)(pa + 8);
            const unsigned short* pb = &sAb[lmod][ks + khalf * 16];
            bfr.q[0] = *(const u32x4*)(pb);
            bfr.q[1] = *(const u32x4*)(pb + 8);
            acc = __builtin_amdgcn_wmma_f32_16x16x32_bf16(
                false, af.v, false, bfr.v, (short)0, acc, false, false);
        }
        __syncthreads();
    }

    #pragma unroll
    for (int r = 0; r < 8; ++r) {
        int row = rowBase + r + khalf * 8;
        U[(size_t)row * 16 + lmod] = acc[r];
    }
}

// ---------------------------------------------------------------------------
// Sequential scan (in place): H[t] = tanh(U[t] + H[t-1]) per (batch, state)
// 64 independent scalar chains; loads independent of h -> pipelined by unroll.
// ---------------------------------------------------------------------------
__global__ void scan_kernel(float* __restrict__ U)
{
    int tid = threadIdx.x;           // 64 threads
    int b = tid >> 4, j = tid & 15;
    size_t base = ((size_t)b * LL) * 16 + j;
    float h = 0.0f;
    #pragma unroll 8
    for (int t = 0; t < LL; ++t) {
        float u = U[base + (size_t)t * 16];
        h = tanhf(u + h);
        U[base + (size_t)t * 16] = h;
    }
}

// ---------------------------------------------------------------------------
// Gating: y = sigmoid(z) * (H @ C^T + Dp * x_conv); y overwrites z half of xz
// ---------------------------------------------------------------------------
__global__ __launch_bounds__(256)
void gate_kernel(const float* __restrict__ H,     // [M,16] (scan output)
                 const float* __restrict__ Cmat,  // [E,16]
                 const float* __restrict__ Dp,    // [E]
                 const unsigned short* __restrict__ xc,
                 unsigned short* __restrict__ xz) // read z / write y in place
{
    __shared__ float hs[16];
    size_t row = blockIdx.x;
    if (threadIdx.x < 16) hs[threadIdx.x] = H[row * 16 + threadIdx.x];
    __syncthreads();
    for (int e = threadIdx.x; e < EE; e += 256) {
        float x = bf2f(xc[row * EE + e]);
        float s = Dp[e] * x;
        #pragma unroll
        for (int j = 0; j < 16; ++j) s += hs[j] * Cmat[(size_t)e * 16 + j];
        float z = bf2f(xz[row * NN2E + EE + e]);
        float y = s * (1.0f / (1.0f + __expf(-z)));
        xz[row * NN2E + EE + e] = f2bf(y);
    }
}

// ---------------------------------------------------------------------------
// Launch
// ---------------------------------------------------------------------------
extern "C" void kernel_launch(void* const* d_in, const int* in_sizes, int n_in,
                              void* d_out, int out_size, void* d_ws, size_t ws_size,
                              hipStream_t stream)
{
    const float* x      = (const float*)d_in[0];  // [B,L,DM]
    const float* W_in   = (const float*)d_in[1];  // [DM, 2E]
    const float* b_in   = (const float*)d_in[2];  // [2E]
    const float* conv_w = (const float*)d_in[3];  // [E,1,4]
    const float* conv_b = (const float*)d_in[4];  // [E]
    const float* Amat   = (const float*)d_in[5];  // [E,16]
    const float* Cmat   = (const float*)d_in[6];  // [E,16]
    const float* Dp     = (const float*)d_in[7];  // [E]
    const float* W_out  = (const float*)d_in[8];  // [E,DM]
    const float* b_out  = (const float*)d_in[9];  // [DM]
    float* out          = (float*)d_out;          // [B,L,DM]

    // Workspace layout:
    //   xz    : [MTOT][4096] bf16 (x_proj | z, z later overwritten by y) 128 MB
    //   xc    : [MTOT][2048] bf16                                         64 MB
    //   U     : [MTOT][16]   f32  (scan in place -> H)                     1 MB
    //   xb    : [MTOT][1024] bf16 (x converted)                           32 MB
    //   WtIn  : [4096][1024] bf16 (W_in  transposed)                       8 MB
    //   WtOut : [1024][2048] bf16 (W_out transposed)                       4 MB
    unsigned short* xz    = (unsigned short*)d_ws;
    unsigned short* xc    = xz + (size_t)MTOT * NN2E;
    float*          U     = (float*)(xc + (size_t)MTOT * EE);
    unsigned short* xb    = (unsigned short*)(U + (size_t)MTOT * 16);
    unsigned short* WtIn  = xb + (size_t)MTOT * DM;
    unsigned short* WtOut = WtIn + (size_t)NN2E * DM;

    // 0) prepasses: bf16 convert of x; bf16 transpose of both weight matrices
    convert_bf16_kernel<<<(size_t)MTOT * DM / 256, 256, 0, stream>>>(x, xb);
    transpose_bf16_kernel<<<dim3(DM / 32, NN2E / 32), dim3(32, 8), 0, stream>>>(
        W_in, WtIn, DM, NN2E);
    transpose_bf16_kernel<<<dim3(EE / 32, DM / 32), dim3(32, 8), 0, stream>>>(
        W_out, WtOut, EE, DM);

    // 1) in_proj GEMM (async-staged bf16 WMMA):  xz = x @ W_in + b_in
    gemm_bf16_wmma_async<true><<<dim3(MTOT / BM, NN2E / BN), 256, 0, stream>>>(
        xb, WtIn, b_in, (void*)xz, DM, DM, NN2E);

    // 2) causal depthwise conv on x_proj half
    conv_kernel<<<(size_t)MTOT * EE / 256, 256, 0, stream>>>(xz, conv_w, conv_b, xc);

    // 3) U = x_conv @ A  (bf16 WMMA, N=16)
    xA_gemm_wmma<<<MTOT / 128, 256, 0, stream>>>(xc, Amat, U);

    // 4) sequential scan H[t] = tanh(U[t] + H[t-1]) (in place)
    scan_kernel<<<1, 64, 0, stream>>>(U);

    // 5) gating: y = sigmoid(z) * (H@C^T + Dp*x_conv), in place over z
    gate_kernel<<<MTOT, 256, 0, stream>>>(U, Cmat, Dp, xc, xz);

    // 6) out_proj GEMM (async-staged bf16 WMMA): out = y @ W_out + b_out
    gemm_bf16_wmma_async<false><<<dim3(MTOT / BM, DM / BN), 256, 0, stream>>>(
        xz + EE, WtOut, b_out, (void*)out, EE, NN2E, DM);
}
